// PreprocessNetwork_57131654971939
// MI455X (gfx1250) — compile-verified, matrix operand order checked by reading, not run on hardware
//
#include <hip/hip_runtime.h>

// CDNA5 (gfx1250) fused kernel: per-neighbor 2->64 3x3 conv (WMMA bf16) + ReLU,
// 1x1 64->1 conv + sigmoid (in-register reduction + lane-half shuffle),
// neighbor weighting, final 7->7 3x3 conv. One pass over x, one write of out.
// bf16 packing done with v_perm_b32 (1 VALU op per 2 elements) instead of
// scalar shift/round chains, to shrink the serial B-fragment build between
// WMMAs.

typedef __attribute__((ext_vector_type(16))) __bf16 v16bf;
typedef __attribute__((ext_vector_type(8)))  float  v8f;

#define TILE   16            // output tile (16x16)
#define WREG   18            // weighted-map region (tile + 1 halo each side)
#define REG_IN 20            // staged input region (tile + 2 halo each side)
#define STR_IN 22            // padded LDS stride for input
#define STR_WT 20            // padded LDS stride for weighted maps
#define NPIX   (WREG * WREG) // 324 weighted pixels
#define NGRP   ((NPIX + 15) / 16) // 21 groups of 16 pixels

union BPack { unsigned int u[8]; v16bf v; };

// Pack two floats into one dword of two bf16 (truncating): one v_perm_b32.
__device__ __forceinline__ unsigned int pack_bf16x2(float lo, float hi) {
  return __builtin_amdgcn_perm(__float_as_uint(hi), __float_as_uint(lo), 0x07060302u);
}

__global__ __launch_bounds__(256)
void preproc_fused_kernel(const float* __restrict__ x,
                          const float* __restrict__ W1,
                          const float* __restrict__ b1,
                          const float* __restrict__ W2,
                          const float* __restrict__ b2,
                          const float* __restrict__ Wm,
                          const float* __restrict__ bm,
                          float* __restrict__ out)
{
  __shared__ float s_in[7][REG_IN][STR_IN];  // 7 frames, 20x20 (+pad) ~12.3 KB
  __shared__ float s_wt[6][WREG][STR_WT];    // 6 weighted maps, 18x18 (+pad) ~8.6 KB

  const int b    = blockIdx.z;
  const int Y0   = blockIdx.y * TILE;
  const int X0   = blockIdx.x * TILE;
  const int tid  = threadIdx.x;
  const int lane = tid & 31;
  const int wave = tid >> 5;

  // ---- Stage 7 input frames (20x20, zero-padded at image borders) into LDS.
  // Zero padding makes out-of-image "weighted" halo values exactly 0
  // (sim * 0), matching SAME-conv semantics of the final conv for free.
  const float* xb = x + (size_t)b * 7u * 256u * 256u;
  for (int i = tid; i < 7 * REG_IN * STR_IN; i += 256) {
    int c  = i / (REG_IN * STR_IN);
    int r  = i % (REG_IN * STR_IN);
    int iy = r / STR_IN;
    int ix = r % STR_IN;
    int gy = Y0 - 2 + iy;
    int gx = X0 - 2 + ix;
    float v = 0.f;
    if (ix < REG_IN && gy >= 0 && gy < 256 && gx >= 0 && gx < 256)
      v = xb[c * 65536 + gy * 256 + gx];
    s_in[c][iy][ix] = v;
  }
  __syncthreads();

  const int lhalf = lane >> 4;   // 0: lanes 0-15, 1: lanes 16-31
  const int lmod  = lane & 15;
  const int kbase = lhalf * 8;   // K sub-range per lane half (ISA 16-bit A/B layout)

  for (int n = 0; n < 6; ++n) {
    const int cc  = (n < 3) ? n : n + 1;   // neighbor channel in x
    const int ch0 = (n < 3) ? 3 : cc;      // pair channel 0
    const int ch1 = (n < 3) ? cc : 3;      // pair channel 1

    // ---- Per-neighbor constants: A fragments (64x18 weights, K padded to 32),
    // bias-initialized accumulators, and 1x1-conv weights laid out to match the
    // C/D fragment (lane half owns channels r vs r+8).
    const float* w1n = W1 + n * 1152;      // 64*2*3*3
    v16bf afrag[4];
    v8f   bias[4];
    float w2r[4][8];
    for (int t = 0; t < 4; ++t) {
      const int m = t * 16 + lmod;         // A-matrix row (output channel)
      float fv[16];
      for (int e = 0; e < 16; ++e) {
        const int k = (e < 8) ? (kbase + e) : (16 + kbase + (e - 8));
        fv[e] = (k < 18) ? w1n[m * 18 + k] : 0.f;
      }
      BPack ap;
      for (int j = 0; j < 8; ++j)
        ap.u[j] = pack_bf16x2(fv[2 * j], fv[2 * j + 1]);
      afrag[t] = ap.v;

      v8f bv;
      for (int r = 0; r < 8; ++r) {
        const int ch = t * 16 + r + lhalf * 8;
        bv[r]     = b1[n * 64 + ch];
        w2r[t][r] = W2[n * 64 + ch];
      }
      bias[t] = bv;
    }
    const float b2n = b2[n];

    // ---- Weighted-map region processed as flat groups of 16 pixels per wave.
    for (int g = wave; g < NGRP; g += 8) {
      const int  p     = g * 16 + lmod;
      const bool valid = (p < NPIX);
      const int  pp    = valid ? p : 0;
      const int  wy    = pp / WREG;
      const int  wx    = pp % WREG;

      // B fragment: im2col patch for pixel column N=lmod (K = ch*9 + dy*3 + dx).
      float fv[16];
      for (int e = 0; e < 16; ++e) {
        const int k = (e < 8) ? (kbase + e) : (16 + kbase + (e - 8));
        float v = 0.f;
        if (k < 18) {
          const int ci = k / 9;
          const int r9 = k % 9;
          v = s_in[ci ? ch1 : ch0][wy + r9 / 3][wx + r9 % 3];
        }
        fv[e] = v;
      }
      BPack bp;
      for (int j = 0; j < 8; ++j)
        bp.u[j] = pack_bf16x2(fv[2 * j], fv[2 * j + 1]);

      v8f c0 = bias[0], c1 = bias[1], c2 = bias[2], c3 = bias[3];
      c0 = __builtin_amdgcn_wmma_f32_16x16x32_bf16(false, afrag[0], false, bp.v, (short)0, c0, false, false);
      c1 = __builtin_amdgcn_wmma_f32_16x16x32_bf16(false, afrag[1], false, bp.v, (short)0, c1, false, false);
      c2 = __builtin_amdgcn_wmma_f32_16x16x32_bf16(false, afrag[2], false, bp.v, (short)0, c2, false, false);
      c3 = __builtin_amdgcn_wmma_f32_16x16x32_bf16(false, afrag[3], false, bp.v, (short)0, c3, false, false);

      // ReLU + 1x1 conv: reduce the 64 channels held across 4 tiles x 8 VGPRs,
      // then combine the two lane halves (same pixel N, channels r vs r+8).
      float part = 0.f;
      for (int r = 0; r < 8; ++r) {
        part += w2r[0][r] * fmaxf(c0[r], 0.f);
        part += w2r[1][r] * fmaxf(c1[r], 0.f);
        part += w2r[2][r] * fmaxf(c2[r], 0.f);
        part += w2r[3][r] * fmaxf(c3[r], 0.f);
      }
      part += __shfl_xor(part, 16, 32);
      const float sim = 1.f / (1.f + __expf(-(part + b2n)));
      const float wv  = sim * s_in[cc][wy + 1][wx + 1];
      if (valid && lhalf == 0)
        s_wt[n][wy][wx] = wv;
    }
  }
  __syncthreads();

  // ---- Final 7->7 3x3 conv: one thread per output pixel, 7 channels each.
  {
    const int oy = tid >> 4;
    const int ox = tid & 15;
    float acc[7];
    for (int co = 0; co < 7; ++co) acc[co] = bm[co];
    for (int ci = 0; ci < 7; ++ci)
      for (int dy = 0; dy < 3; ++dy)
        for (int dx = 0; dx < 3; ++dx) {
          const float v = (ci == 3)
              ? s_in[3][oy + dy + 1][ox + dx + 1]
              : s_wt[(ci < 3) ? ci : ci - 1][oy + dy][ox + dx];
          for (int co = 0; co < 7; ++co)
            acc[co] += Wm[((co * 7 + ci) * 3 + dy) * 3 + dx] * v;  // uniform idx -> s_load
        }
    const size_t obase = (size_t)b * 7u * 65536u + (size_t)(Y0 + oy) * 256u + (size_t)(X0 + ox);
    for (int co = 0; co < 7; ++co)
      out[obase + (size_t)co * 65536u] = acc[co];
  }
}

extern "C" void kernel_launch(void* const* d_in, const int* in_sizes, int n_in,
                              void* d_out, int out_size, void* d_ws, size_t ws_size,
                              hipStream_t stream) {
  (void)in_sizes; (void)n_in; (void)d_ws; (void)ws_size; (void)out_size;
  const float* x  = (const float*)d_in[0];
  const float* W1 = (const float*)d_in[1];
  const float* b1 = (const float*)d_in[2];
  const float* W2 = (const float*)d_in[3];
  const float* b2 = (const float*)d_in[4];
  const float* Wm = (const float*)d_in[5];
  const float* bm = (const float*)d_in[6];
  float* out = (float*)d_out;

  dim3 grid(256 / TILE, 256 / TILE, 8);  // 16 x 16 tiles x B=8
  dim3 block(256);                       // 8 wave32 per block
  preproc_fused_kernel<<<grid, block, 0, stream>>>(x, W1, b1, W2, b2, Wm, bm, out);
}
